// PointCNN_partseg_79190607004309
// MI455X (gfx1250) — matrix operand order repaired and checked by reading, not compiled, not run on hardware
//
#include <hip/hip_runtime.h>
#include <math.h>

// ---------------------------------------------------------------------------
// PointCNN part-seg forward for gfx1250 (MI455X).
// All dense layers go through a generic bf16-WMMA GEMM (v_wmma_f32_16x16x32_bf16,
// f32 accumulate).  kNN / gathers / per-point X-transform application stay VALU.
// ---------------------------------------------------------------------------

typedef __bf16 bf16_t;
typedef __attribute__((ext_vector_type(16))) __bf16 v16bf;
typedef __attribute__((ext_vector_type(8)))  float  v8f;

#define WG 256
#define BM 128
#define BN 32
#define BK 64
#define NA ((BM * BK) / WG)   // 32 A elements per thread per chunk
#define NB ((BK * BN) / WG)   // 8  B elements per thread per chunk

__device__ __forceinline__ float eluf(float x) { return x > 0.f ? x : __expf(x) - 1.f; }

// A-tile K-permutation so each lane's fragment is one contiguous 32B LDS read.
// Fragment order for a 16x32 bf16 A tile (ISA 7.12.2): half0 reads K
// {0..7,16..23}, half1 reads K {8..15,24..31}.  Storing K at position
// pos = ((c>>4)<<3) | (((c>>3)&1)<<4) | (c&7)  makes both halves contiguous.
__device__ __forceinline__ int aperm(int c)
{
  int c5 = c & 31;
  int pos = (((c5 >> 4) << 3) | (((c5 >> 3) & 1) << 4) | (c5 & 7));
  return (c & 32) + pos;
}

__device__ __forceinline__ void wmma_quad(const bf16_t (*sA)[BK], const bf16_t (*sB)[BK],
                                          int ar, int half, int l15,
                                          v8f& acc0, v8f& acc1)
{
  const v16bf af0 = *(const v16bf*)&sA[ar][half * 16];
  const v16bf af1 = *(const v16bf*)&sA[ar][32 + half * 16];
  const v16bf b00 = *(const v16bf*)&sB[l15][half * 16];
  const v16bf b01 = *(const v16bf*)&sB[16 + l15][half * 16];
  const v16bf b10 = *(const v16bf*)&sB[l15][32 + half * 16];
  const v16bf b11 = *(const v16bf*)&sB[16 + l15][32 + half * 16];
  acc0 = __builtin_amdgcn_wmma_f32_16x16x32_bf16(false, af0, false, b00,
                                                 (short)0, acc0, false, false);
  acc1 = __builtin_amdgcn_wmma_f32_16x16x32_bf16(false, af0, false, b01,
                                                 (short)0, acc1, false, false);
  acc0 = __builtin_amdgcn_wmma_f32_16x16x32_bf16(false, af1, false, b10,
                                                 (short)0, acc0, false, false);
  acc1 = __builtin_amdgcn_wmma_f32_16x16x32_bf16(false, af1, false, b11,
                                                 (short)0, acc1, false, false);
}

// ---------------------------------------------------------------------------
// Generic GEMM: C[M,N] = act(A[M,Kd] @ B[Kd,N] + bias[N])
// A, B f32 row-major; staged through double-buffered LDS as bf16 in natural k
// order.  Block = 8 waves; wave w owns rows [w*16,w*16+16) x cols [0,32) of
// the 128x32 block tile; 4 WMMAs per K-chunk (2 K-chunks x 2 N-tiles) with
// A-fragment reuse.  Pipeline: store regs->LDS[p]; barrier; issue next
// chunk's global loads (batched, one wait); WMMA on LDS[p].  Hot path is
// clamp-only (no selects); the partial K-chunk runs as a separate tail.
// ---------------------------------------------------------------------------
__global__ __launch_bounds__(256) void k_gemm(const float* __restrict__ A,
                                              const float* __restrict__ B,
                                              const float* __restrict__ bias,
                                              float* __restrict__ C,
                                              int M, int Kd, int N, int act)
{
  __shared__ __align__(32) bf16_t sA[2][BM][BK];  // 32 KB
  __shared__ __align__(32) bf16_t sB[2][BN][BK];  // 8 KB
  const int tid  = threadIdx.x;
  const int wave = tid >> 5;
  const int lane = tid & 31;
  const int half = lane >> 4;
  const int l15  = lane & 15;
  const long m0  = (long)blockIdx.y * BM;
  const int  n0  = blockIdx.x * BN;
  const int  ar  = wave * 16 + l15;

  v8f acc0, acc1;
#pragma unroll
  for (int i = 0; i < 8; ++i) { acc0[i] = 0.f; acc1[i] = 0.f; }

  float ra[NA], rb[NB];

  // batched global loads for one full K-chunk (clamp-only; OOB rows/cols feed
  // outputs that the epilogue discards)
  auto load_full = [&](int k0) {
#pragma unroll
    for (int t = 0; t < NA; ++t) {
      int i = tid + t * WG;
      int r = i >> 6, c = i & (BK - 1);
      long mm = m0 + r;
      long mc = mm < M ? mm : (long)(M - 1);
      ra[t] = A[mc * (long)Kd + k0 + c];
    }
#pragma unroll
    for (int t = 0; t < NB; ++t) {
      int i = tid + t * WG;
      int r = i >> 5, c = i & (BN - 1);
      int nn = n0 + c;
      int nc = nn < N ? nn : (N - 1);
      rb[t] = B[(long)(k0 + r) * N + nc];
    }
  };
  // convert + scatter regs into LDS buffer p (A in natural k order, B transposed)
  auto store_tile = [&](int p) {
#pragma unroll
    for (int t = 0; t < NA; ++t) {
      int i = tid + t * WG;
      int r = i >> 6, c = i & (BK - 1);
      sA[p][r][c] = (bf16_t)ra[t];
    }
#pragma unroll
    for (int t = 0; t < NB; ++t) {
      int i = tid + t * WG;
      int r = i >> 5, c = i & (BN - 1);
      sB[p][c][r] = (bf16_t)rb[t];
    }
  };

  const int kfull  = Kd & ~(BK - 1);
  const int nfull  = kfull / BK;

  // ---- pipelined hot loop over full K-chunks ----
  if (nfull > 0) {
    load_full(0);
    int p = 0;
    for (int ch = 0; ch < nfull; ++ch) {
      store_tile(p);
      __syncthreads();
      if (ch + 1 < nfull) load_full((ch + 1) * BK);   // overlaps with WMMA below
      wmma_quad(sA[p], sB[p], ar, half, l15, acc0, acc1);
      p ^= 1;
    }
  }

  // ---- tail: partial K-chunk, zero-pad the K dimension ----
  if (kfull < Kd) {
    int k0 = kfull;
    __syncthreads();   // buffer 0 may still be being read by other waves
#pragma unroll
    for (int t = 0; t < NA; ++t) {
      int i = tid + t * WG;
      int r = i >> 6, c = i & (BK - 1);
      long mm = m0 + r; int kk = k0 + c;
      long mc = mm < M ? mm : (long)(M - 1);
      int  kc = kk < Kd ? kk : (Kd - 1);
      float v = A[mc * (long)Kd + kc];
      v = (kk < Kd) ? v : 0.f;
      sA[0][r][c] = (bf16_t)v;
    }
#pragma unroll
    for (int t = 0; t < NB; ++t) {
      int i = tid + t * WG;
      int r = i >> 5, c = i & (BN - 1);
      int kk = k0 + r, nn = n0 + c;
      int kc = kk < Kd ? kk : (Kd - 1);
      int nc = nn < N ? nn : (N - 1);
      float v = B[(long)kc * N + nc];
      v = (kk < Kd) ? v : 0.f;
      sB[0][c][r] = (bf16_t)v;
    }
    __syncthreads();
    wmma_quad(sA[0], sB[0], ar, half, l15, acc0, acc1);
  }

  // ---- epilogue: bias + activation + store both N tiles ----
  const int nnA = n0 + l15;
  const int nnB = n0 + 16 + l15;
  float bvA = (bias != nullptr && nnA < N) ? bias[nnA] : 0.f;
  float bvB = (bias != nullptr && nnB < N) ? bias[nnB] : 0.f;
#pragma unroll
  for (int i = 0; i < 8; ++i) {
    long mm = m0 + wave * 16 + half * 8 + i;
    if (mm < M) {
      if (nnA < N) {
        float v = acc0[i] + bvA;
        if (act) v = eluf(v);
        C[mm * (long)N + nnA] = v;
      }
      if (nnB < N) {
        float v = acc1[i] + bvB;
        if (act) v = eluf(v);
        C[mm * (long)N + nnB] = v;
      }
    }
  }
}

// ---------------------------------------------------------------------------
// Small data-movement / irregular kernels
// ---------------------------------------------------------------------------

// x (N,3,M) -> pts (N,M,3)
__global__ void k_transpose_in(const float* __restrict__ x, float* __restrict__ pts,
                               int Nb, int M)
{
  long i = (long)blockIdx.x * blockDim.x + threadIdx.x;
  long tot = (long)Nb * M * 3;
  if (i >= tot) return;
  int d = (int)(i % 3);
  long nm = i / 3;
  int m = (int)(nm % M);
  int n = (int)(nm / M);
  pts[i] = x[((long)n * 3 + d) * M + m];
}

// rep (N,P,3) = pts[:, sidx, :]
__global__ void k_gather_pts(const float* __restrict__ pts, const int* __restrict__ sidx,
                             float* __restrict__ rep, int Nb, int M, int P)
{
  long i = (long)blockIdx.x * blockDim.x + threadIdx.x;
  long tot = (long)Nb * P * 3;
  if (i >= tot) return;
  int d = (int)(i % 3);
  long np = i / 3;
  int p = (int)(np % P);
  int n = (int)(np / P);
  rep[i] = pts[((long)n * M + sidx[p]) * 3 + d];
}

// brute-force kNN: keep K+1 smallest sq-distances, drop the nearest (self)
__global__ void k_knn(const float* __restrict__ rep, const float* __restrict__ pts,
                      int* __restrict__ idx, int Nb, int P, int M, int K)
{
  long t = (long)blockIdx.x * blockDim.x + threadIdx.x;
  if (t >= (long)Nb * P) return;
  int n = (int)(t / P);
  const float* r = rep + t * 3;
  float rx = r[0], ry = r[1], rz = r[2];
  const float* pb = pts + (long)n * M * 3;
  float bd[17];
  int   bi[17];
  int KP = K + 1;
  for (int i = 0; i < KP; ++i) { bd[i] = 3.4e38f; bi[i] = 0; }
  for (int m = 0; m < M; ++m) {
    float dx = pb[m * 3 + 0] - rx;
    float dy = pb[m * 3 + 1] - ry;
    float dz = pb[m * 3 + 2] - rz;
    float d = dx * dx + dy * dy + dz * dz;
    if (d < bd[KP - 1]) {
      int j = KP - 1;
      while (j > 0 && bd[j - 1] > d) { bd[j] = bd[j - 1]; bi[j] = bi[j - 1]; --j; }
      bd[j] = d; bi[j] = m;
    }
  }
  for (int k = 1; k < KP; ++k) idx[t * K + (k - 1)] = bi[k];
}

// loc (N,P,K,3) = pts[idx] - rep
__global__ void k_loc(const float* __restrict__ pts, const float* __restrict__ rep,
                      const int* __restrict__ idx, float* __restrict__ loc,
                      int Nb, int P, int M, int K)
{
  long i = (long)blockIdx.x * blockDim.x + threadIdx.x;
  long tot = (long)Nb * P * K * 3;
  if (i >= tot) return;
  int d = (int)(i % 3);
  long npk = i / 3;
  long np = npk / K;
  int n = (int)(np / P);
  loc[i] = pts[((long)n * M + idx[npk]) * 3 + d] - rep[np * 3 + d];
}

// xw (3,K,KK) -> xwt (K*3, KK) with row index k*3+d (matches loc's (k,d) flatten)
__global__ void k_xwt(const float* __restrict__ xw, float* __restrict__ xwt, int K, int KK)
{
  long i = (long)blockIdx.x * blockDim.x + threadIdx.x;
  long tot = (long)3 * K * KK;
  if (i >= tot) return;
  int o  = (int)(i % KK);
  int dk = (int)(i / KK);
  int k  = dk % K;
  int d  = dk / K;
  xwt[((long)k * 3 + d) * KK + o] = xw[i];
}

// Per rep-point: fx = X(KxK) @ fcat(K x cx) then depthwise conv over K:
//   dw[np, c*dm+m] = sum_k fx[k,c]*dww[c,m,k] + dwb
// fcat column c is f2[np,:,c] for c<cmid else gathered fhalf[idx[np,:], c-cmid].
__global__ __launch_bounds__(128) void k_xconv_dw(const float* __restrict__ X,
                                                  const float* __restrict__ f2,
                                                  const float* __restrict__ fhalf,
                                                  const int* __restrict__ idx,
                                                  const float* __restrict__ dww,
                                                  const float* __restrict__ dwb,
                                                  float* __restrict__ dw,
                                                  int P, int Mpts, int K,
                                                  int cmid, int chalf, int dm)
{
  long np = blockIdx.x;              // n*P + p
  int n = (int)(np / P);
  int cx = cmid + chalf;
  int KK = K * K;
  __shared__ float sX[256];
  __shared__ int   sIdx[16];
  for (int i = threadIdx.x; i < KK; i += blockDim.x) sX[i] = X[np * KK + i];
  for (int i = threadIdx.x; i < K;  i += blockDim.x) sIdx[i] = idx[np * K + i];
  __syncthreads();

  for (int c = threadIdx.x; c < cx; c += blockDim.x) {
    float col[16];
    if (c < cmid) {
      for (int k = 0; k < K; ++k) col[k] = f2[(np * K + k) * (long)cmid + c];
    } else {
      int cc = c - cmid;
      for (int k = 0; k < K; ++k)
        col[k] = fhalf[((long)n * Mpts + sIdx[k]) * (long)chalf + cc];
    }
    float fx[16];
    for (int i = 0; i < K; ++i) {
      float s = 0.f;
      for (int j = 0; j < K; ++j) s += sX[i * K + j] * col[j];
      fx[i] = s;
    }
    for (int m = 0; m < dm; ++m) {
      const float* w = dww + ((long)c * dm + m) * K;
      float s = 0.f;
      for (int k = 0; k < K; ++k) s += fx[k] * w[k];
      int o = c * dm + m;
      dw[np * (long)(cx * dm) + o] = s + dwb[o];
    }
  }
}

// c (NP, c1+c2) = concat(a (NP,c1), b (NP,c2))
__global__ void k_concat(const float* __restrict__ a, const float* __restrict__ b,
                         float* __restrict__ c, long NP, int c1, int c2)
{
  long i = (long)blockIdx.x * blockDim.x + threadIdx.x;
  int ct = c1 + c2;
  long tot = NP * ct;
  if (i >= tot) return;
  int ch = (int)(i % ct);
  long np = i / ct;
  c[i] = (ch < c1) ? a[np * c1 + ch] : b[np * c2 + (ch - c1)];
}

// f (N,P,C) -> out (N,C,P)
__global__ void k_transpose_out(const float* __restrict__ f, float* __restrict__ out,
                                int Nb, int P, int C)
{
  long i = (long)blockIdx.x * blockDim.x + threadIdx.x;
  long tot = (long)Nb * P * C;
  if (i >= tot) return;
  int p = (int)(i % P);
  long ncp = i / P;
  int c = (int)(ncp % C);
  int n = (int)(ncp / C);
  out[i] = f[((long)n * P + p) * C + c];
}

// ---------------------------------------------------------------------------
// Host-side orchestration
// ---------------------------------------------------------------------------

struct DenseP { const float *b, *w; };
struct Layer {
  DenseP d1, d2, dense, fuse, xd1, xd2;
  const float *dwb, *dww, *pww, *xb, *xw;
};

// JAX pytree flatten order: dict keys sorted alphabetically at each level.
// Layer keys: d1,d2,dense,dwb,dww,[fuse],pww,xb,xd1,xd2,xw ; Dense: b,w
static int load_layer(void* const* din, int i, Layer& L, bool has_fuse)
{
  L.d1.b    = (const float*)din[i++]; L.d1.w    = (const float*)din[i++];
  L.d2.b    = (const float*)din[i++]; L.d2.w    = (const float*)din[i++];
  L.dense.b = (const float*)din[i++]; L.dense.w = (const float*)din[i++];
  L.dwb     = (const float*)din[i++];
  L.dww     = (const float*)din[i++];
  if (has_fuse) { L.fuse.b = (const float*)din[i++]; L.fuse.w = (const float*)din[i++]; }
  else          { L.fuse.b = nullptr;                L.fuse.w = nullptr; }
  L.pww     = (const float*)din[i++];
  L.xb      = (const float*)din[i++];
  L.xd1.b   = (const float*)din[i++]; L.xd1.w = (const float*)din[i++];
  L.xd2.b   = (const float*)din[i++]; L.xd2.w = (const float*)din[i++];
  L.xw      = (const float*)din[i++];
  return i;
}

struct Scratch {
  float *fhalf, *loc, *f1, *f2, *X1, *X2, *dw, *xwt;
  int* idx;
};

static inline void gemm(const float* A, const float* B, const float* bias, float* C,
                        long M, int Kd, int N, int act, hipStream_t s)
{
  dim3 g((unsigned)((N + BN - 1) / BN), (unsigned)((M + BM - 1) / BM));
  k_gemm<<<g, WG, 0, s>>>(A, B, bias, C, (int)M, Kd, N, act);
}

static inline unsigned gblk(long tot, int bs) { return (unsigned)((tot + bs - 1) / bs); }

// One full PointCNN XConv layer: dense lift + kNN + coord lift + X-transform
// + (X @ fcat -> depthwise) + pointwise.  outF is (N,P,cout).
static void run_xconv(const Layer& L, const float* rep, const float* pts,
                      const float* ftsin, int Nb, int M, int P, int K,
                      int cin, int cout, float* outF, const Scratch& S, hipStream_t st)
{
  int cmid = cout / 4, chalf = cout / 2;
  int dm = (cout + cin - 1) / cin; if (dm > 4) dm = 4;
  int cx = cmid + chalf, KK = K * K;
  long NM = (long)Nb * M, NP = (long)Nb * P, NPK = NP * K;

  // fts = elu(ftsin @ dense.w + b)           (N,M,chalf)
  gemm(ftsin, L.dense.w, L.dense.b, S.fhalf, NM, cin, chalf, 1, st);
  // idx = knn(rep, pts, K)                   (N,P,K)
  k_knn<<<gblk(NP, 128), 128, 0, st>>>(rep, pts, S.idx, Nb, P, M, K);
  // loc = pts[idx] - rep                     (N,P,K,3)
  k_loc<<<gblk(NPK * 3, 256), 256, 0, st>>>(pts, rep, S.idx, S.loc, Nb, P, M, K);
  // f = elu(d2(elu(d1(loc))))                (N,P,K,cmid)
  gemm(S.loc, L.d1.w, L.d1.b, S.f1, NPK, 3, cmid, 1, st);
  gemm(S.f1, L.d2.w, L.d2.b, S.f2, NPK, cmid, cmid, 1, st);
  // X = xd2(xd1(elu(loc @ xw + xb)))         (N,P,K*K)
  k_xwt<<<gblk((long)3 * K * KK, 256), 256, 0, st>>>(L.xw, S.xwt, K, KK);
  gemm(S.loc, S.xwt, L.xb, S.X1, NP, 3 * K, KK, 1, st);
  gemm(S.X1, L.xd1.w, L.xd1.b, S.X2, NP, KK, KK, 1, st);
  gemm(S.X2, L.xd2.w, L.xd2.b, S.X1, NP, KK, KK, 0, st);
  // dw = depthwise(X @ concat(f, gather(fts))) + dwb   (N,P,cx*dm)
  k_xconv_dw<<<dim3((unsigned)NP), 128, 0, st>>>(S.X1, S.f2, S.fhalf, S.idx,
                                                 L.dww, L.dwb, S.dw,
                                                 P, M, K, cmid, chalf, dm);
  // out = elu(dw @ pww)                      (N,P,cout)
  gemm(S.dw, L.pww, nullptr, outF, NP, cx * dm, cout, 1, st);
}

extern "C" void kernel_launch(void* const* d_in, const int* in_sizes, int n_in,
                              void* d_out, int out_size, void* d_ws, size_t ws_size,
                              hipStream_t stream)
{
  (void)in_sizes; (void)n_in; (void)out_size; (void)ws_size;

  // --- unpack params (sorted pytree order: dec[0..3], enc[0..3], sidx1-3, x)
  Layer dec[4], enc[4];
  int pi = 0;
  for (int i = 0; i < 4; ++i) pi = load_layer(d_in, pi, dec[i], true);
  for (int i = 0; i < 4; ++i) pi = load_layer(d_in, pi, enc[i], false);
  const int*   sidx1 = (const int*)d_in[pi++];
  const int*   sidx2 = (const int*)d_in[pi++];
  const int*   sidx3 = (const int*)d_in[pi++];
  const float* x     = (const float*)d_in[pi++];

  const int N = 16;

  // --- workspace arena
  char* base = (char*)d_ws;
  size_t off = 0;
  auto alloc = [&](size_t elems) -> float* {
    float* p = (float*)(base + off);
    off += ((elems * sizeof(float) + 255) & ~(size_t)255);
    return p;
  };
  Scratch S;
  S.fhalf = alloc(4194304);          // max N*M*chalf
  S.idx   = (int*)alloc(262144);     // max N*P*K
  S.loc   = alloc(786432);           // max N*P*K*3
  S.f1    = alloc(16777216);         // max N*P*K*cmid
  S.f2    = alloc(16777216);
  S.X1    = alloc(2097152);          // max N*P*K*K
  S.X2    = alloc(2097152);
  S.dw    = alloc(25165824);         // max N*P*cx*dm
  S.xwt   = alloc(16384);            // max 3K * K^2
  float* fd   = alloc(3145728);      // decoder xconv output
  float* cat  = alloc(10027008);     // decoder concat buffer
  float* curA = alloc(3145728);      // decoder fts ping
  float* curB = alloc(3145728);      // decoder fts pong
  float* pts0 = alloc(98304);        // (16,2048,3)
  float* rep1 = alloc(36864);        // (16,768,3)
  float* rep2 = alloc(18432);        // (16,384,3)
  float* rep3 = alloc(6144);         // (16,128,3)
  float* F0   = alloc(8388608);      // (16,2048,256)
  float* F1   = alloc(3145728);      // (16,768,256)
  float* F2   = alloc(3145728);      // (16,384,512)
  float* F3   = alloc(2097152);      // (16,128,1024)

  // --- encoder -------------------------------------------------------------
  k_transpose_in<<<gblk((long)N * 2048 * 3, 256), 256, 0, stream>>>(x, pts0, N, 2048);

  // E0: 3 -> 256, K=8, P=2048 (rep = pts)
  run_xconv(enc[0], pts0, pts0, pts0, N, 2048, 2048, 8, 3, 256, F0, S, stream);
  // E1: 256 -> 256, K=12, P=768
  k_gather_pts<<<gblk((long)N * 768 * 3, 256), 256, 0, stream>>>(pts0, sidx1, rep1, N, 2048, 768);
  run_xconv(enc[1], rep1, pts0, F0, N, 2048, 768, 12, 256, 256, F1, S, stream);
  // E2: 256 -> 512, K=16, P=384
  k_gather_pts<<<gblk((long)N * 384 * 3, 256), 256, 0, stream>>>(rep1, sidx2, rep2, N, 768, 384);
  run_xconv(enc[2], rep2, rep1, F1, N, 768, 384, 16, 256, 512, F2, S, stream);
  // E3: 512 -> 1024, K=16, P=128
  k_gather_pts<<<gblk((long)N * 128 * 3, 256), 256, 0, stream>>>(rep2, sidx3, rep3, N, 384, 128);
  run_xconv(enc[3], rep3, rep2, F2, N, 384, 128, 16, 512, 1024, F3, S, stream);

  const float* ptsArr[4] = { pts0, rep1, rep2, rep3 };
  const float* Fs[4]     = { F0, F1, F2, F3 };
  const int    Parr[4]   = { 2048, 768, 384, 128 };

  // --- decoder -------------------------------------------------------------
  const int decM[4]  = { 128, 128, 384, 768 };   // #points in current cloud
  const int decK[4]  = { 16, 16, 12, 8 };
  const int decCi[4] = { 1024, 1024, 512, 256 };
  const int decCo[4] = { 1024, 512, 256, 50 };
  const int decCl[4] = { 1024, 512, 256, 256 };

  const float* cur    = F3;
  const float* curpts = rep3;
  float* curbuf[2] = { curA, curB };

  for (int i = 0; i < 4; ++i) {
    int ridx = 3 - i;
    const float* repp = ptsArr[ridx];
    const float* repF = Fs[ridx];
    int P = Parr[ridx];
    int M = decM[i];
    int K = decK[i], ci = decCi[i], co = decCo[i], cl = decCl[i];

    run_xconv(dec[i], repp, curpts, cur, N, M, P, K, ci, co, fd, S, stream);

    long NP = (long)N * P;
    k_concat<<<gblk(NP * (co + cl), 256), 256, 0, stream>>>(fd, repF, cat, NP, co, cl);
    gemm(cat, dec[i].fuse.w, dec[i].fuse.b, curbuf[i & 1], NP, co + cl, co, 1, stream);
    cur = curbuf[i & 1];
    curpts = repp;
  }

  // --- output: (N,2048,50) -> (N,50,2048)
  k_transpose_out<<<gblk((long)N * 2048 * 50, 256), 256, 0, stream>>>(
      cur, (float*)d_out, N, 2048, 50);
}
